// NMT_49770081026192
// MI455X (gfx1250) — matrix operand order repaired and checked by reading, not compiled, not run on hardware
//
#include <hip/hip_runtime.h>
#include <math.h>

// ---------------------------------------------------------------------------
// Problem constants
// ---------------------------------------------------------------------------
#define S_ 48
#define B_ 32
#define T_ 50
#define E_ 256
#define H_ 512
#define V_ 32000
#define G_ 2048           // 4*H
#define TD_ 49            // T-1 decoder steps

// ---------------------------------------------------------------------------
// Types for WMMA (gfx1250, wave32)
// ---------------------------------------------------------------------------
typedef __attribute__((ext_vector_type(16))) __bf16          v16bf;
typedef __attribute__((ext_vector_type(8)))  float           v8f;
typedef __attribute__((ext_vector_type(8)))  unsigned short  v8us;
typedef __attribute__((ext_vector_type(16))) unsigned short  v16us;

union FragA { v16bf bf; v8us  us[2]; };
union FragB { v16bf bf; v16us us16;  };

__device__ __forceinline__ unsigned short f32_to_bf16(float f) {
    unsigned int u = __float_as_uint(f);
    unsigned int r = (u + 0x7FFFu + ((u >> 16) & 1u)) >> 16;   // RNE
    return (unsigned short)r;
}
__device__ __forceinline__ float sigm(float x) { return 1.0f / (1.0f + __expf(-x)); }

// ---------------------------------------------------------------------------
// bf16 WMMA GEMM:  C[M,N] = A[M,K] * B[N,K]^T (+bias[N])
//   A,B row-major bf16 (raw ushort), C row-major f32.
//   Requires M%32==0, N%256==0, K%32==0 (true for every call below).
//   Block = 128 threads = 4 waves; wave tile 32x64; block tile 32x256.
// ---------------------------------------------------------------------------
__global__ __launch_bounds__(128)
void k_gemm_bf16(const unsigned short* __restrict__ A, int lda,
                 const unsigned short* __restrict__ B, int ldb,
                 float* __restrict__ C, int ldc,
                 const float* __restrict__ bias,
                 int M, int N, int K)
{
    const int lane = threadIdx.x & 31;
    const int wave = threadIdx.x >> 5;
    const int m0 = blockIdx.y * 32;
    const int n0 = blockIdx.x * 256 + wave * 64;
    if (m0 >= M || n0 >= N) return;

    const int lrow = lane & 15;
    const int lhi  = lane >> 4;

    v8f acc[2][4];
    #pragma unroll
    for (int r = 0; r < 2; ++r)
        #pragma unroll
        for (int i = 0; i < 4; ++i) acc[r][i] = {};

    // A fragment base pointers (16x32, contiguous k order used here; B matches):
    //   lanes 0-15 : v[0..15]=K 0..15
    //   lanes16-31 : v[0..15]=K16..31
    const unsigned short* Arow0 = A + (size_t)(m0 + lrow) * lda + (lhi ? 16 : 0);
    const unsigned short* Arow1 = Arow0 + (size_t)16 * lda;

    // B fragment (32x16): lanes 0-15 hold K0..15, lanes 16-31 hold K16..31
    const unsigned short* Bcol[4];
    #pragma unroll
    for (int i = 0; i < 4; ++i)
        Bcol[i] = B + (size_t)(n0 + 16 * i + lrow) * ldb + (lhi ? 16 : 0);

    for (int kk = 0; kk < K; kk += 32) {
        FragA a0, a1;
        a0.us[0] = *(const v8us*)(Arow0 + kk);
        a0.us[1] = *(const v8us*)(Arow0 + kk + 8);
        a1.us[0] = *(const v8us*)(Arow1 + kk);
        a1.us[1] = *(const v8us*)(Arow1 + kk + 8);
        #pragma unroll
        for (int i = 0; i < 4; ++i) {
            FragB fb;
            fb.us16 = *(const v16us*)(Bcol[i] + kk);
            acc[0][i] = __builtin_amdgcn_wmma_f32_16x16x32_bf16(
                false, a0.bf, false, fb.bf, (short)0, acc[0][i], false, false);
            acc[1][i] = __builtin_amdgcn_wmma_f32_16x16x32_bf16(
                false, a1.bf, false, fb.bf, (short)0, acc[1][i], false, false);
        }
    }

    // C layout: VGPR e -> row m0+16r+8*lhi+e, column = n0+16i+lrow
    #pragma unroll
    for (int i = 0; i < 4; ++i) {
        const int col = n0 + 16 * i + lrow;
        const float bv = bias ? bias[col] : 0.0f;
        #pragma unroll
        for (int r = 0; r < 2; ++r) {
            #pragma unroll
            for (int e = 0; e < 8; ++e) {
                const int row = m0 + 16 * r + 8 * lhi + e;
                C[(size_t)row * ldc + col] = acc[r][i][e] + bv;
            }
        }
    }
}

// ---------------------------------------------------------------------------
// Small utility kernels
// ---------------------------------------------------------------------------
__global__ void k_cvt_bf16(const float* __restrict__ src,
                           unsigned short* __restrict__ dst, int n)
{
    int i = blockIdx.x * blockDim.x + threadIdx.x;
    if (i < n) dst[i] = f32_to_bf16(src[i]);
}

__global__ void k_slice_bf16(const float* __restrict__ src, int src_ld, int scol,
                             unsigned short* __restrict__ dst, int dst_ld, int dcol,
                             int rows, int cols)
{
    int i = blockIdx.x * blockDim.x + threadIdx.x;
    if (i >= rows * cols) return;
    int c = i % cols, r = i / cols;
    dst[(size_t)r * dst_ld + dcol + c] = f32_to_bf16(src[(size_t)r * src_ld + scol + c]);
}

__global__ void k_zero_f32(float* p, int n)
{
    int i = blockIdx.x * blockDim.x + threadIdx.x;
    if (i < n) p[i] = 0.0f;
}
__global__ void k_zero_us(unsigned short* p, int n)
{
    int i = blockIdx.x * blockDim.x + threadIdx.x;
    if (i < n) p[i] = 0;
}

// row r = a*Bd + b ; token = tokens[a*sa + b*sb]
__global__ void k_embed(const int* __restrict__ tokens, int sa, int sb,
                        const float* __restrict__ emb,
                        unsigned short* __restrict__ dst, int A, int Bd, int E)
{
    int i = blockIdx.x * blockDim.x + threadIdx.x;
    if (i >= A * Bd * E) return;
    int e = i % E;
    int r = i / E;
    int b = r % Bd, a = r / Bd;
    int tok = tokens[a * sa + b * sb];
    dst[i] = f32_to_bf16(emb[(size_t)tok * E + e]);
}

// ---------------------------------------------------------------------------
// Encoder LSTM step (masked). gates = h@Whh^T ; xpre_t = x_t@Wih^T
// ---------------------------------------------------------------------------
__global__ void k_lstm_enc_step(const float* __restrict__ gates,
                                const float* __restrict__ xpre_t,
                                const float* __restrict__ bih,
                                const float* __restrict__ bhh,
                                float* __restrict__ h, float* __restrict__ c,
                                unsigned short* __restrict__ h_bf,
                                unsigned short* __restrict__ encout,
                                const int* __restrict__ src_lens,
                                int t, int col_off)
{
    int idx = blockIdx.x * blockDim.x + threadIdx.x;
    if (idx >= B_ * H_) return;
    int b = idx >> 9, j = idx & (H_ - 1);
    bool valid = t < src_lens[b];
    const int g0 = b * G_;
    float gi = gates[g0 + j]            + xpre_t[g0 + j]            + bih[j]            + bhh[j];
    float gf = gates[g0 + H_ + j]       + xpre_t[g0 + H_ + j]       + bih[H_ + j]       + bhh[H_ + j];
    float gg = gates[g0 + 2 * H_ + j]   + xpre_t[g0 + 2 * H_ + j]   + bih[2 * H_ + j]   + bhh[2 * H_ + j];
    float go = gates[g0 + 3 * H_ + j]   + xpre_t[g0 + 3 * H_ + j]   + bih[3 * H_ + j]   + bhh[3 * H_ + j];
    float cold = c[idx], hold = h[idx];
    float cn = sigm(gf) * cold + sigm(gi) * tanhf(gg);
    float hn = sigm(go) * tanhf(cn);
    float hw = valid ? hn : hold;
    float cw = valid ? cn : cold;
    h[idx] = hw; c[idx] = cw;
    h_bf[idx] = f32_to_bf16(hw);
    encout[(size_t)(t * B_ + b) * (2 * H_) + col_off + j] = f32_to_bf16(valid ? hn : 0.0f);
}

// Decoder LSTM step (unmasked). gates = [ctx,h]@Wcomb^T ; xpre_t = x_t@Wx^T
__global__ void k_lstm_dec_step(const float* __restrict__ gates,
                                const float* __restrict__ xpre_t,
                                const float* __restrict__ bih,
                                const float* __restrict__ bhh,
                                float* __restrict__ h, float* __restrict__ c)
{
    int idx = blockIdx.x * blockDim.x + threadIdx.x;
    if (idx >= B_ * H_) return;
    int b = idx >> 9, j = idx & (H_ - 1);
    const int g0 = b * G_;
    float gi = gates[g0 + j]          + xpre_t[g0 + j]          + bih[j]          + bhh[j];
    float gf = gates[g0 + H_ + j]     + xpre_t[g0 + H_ + j]     + bih[H_ + j]     + bhh[H_ + j];
    float gg = gates[g0 + 2*H_ + j]   + xpre_t[g0 + 2*H_ + j]   + bih[2*H_ + j]   + bhh[2*H_ + j];
    float go = gates[g0 + 3*H_ + j]   + xpre_t[g0 + 3*H_ + j]   + bih[3*H_ + j]   + bhh[3*H_ + j];
    float cn = sigm(gf) * c[idx] + sigm(gi) * tanhf(gg);
    float hn = sigm(go) * tanhf(cn);
    c[idx] = cn; h[idx] = hn;
}

// catc = bf16([c_f, c_b])  (B x 2H)
__global__ void k_concat_c(const float* __restrict__ cf, const float* __restrict__ cb,
                           unsigned short* __restrict__ catc)
{
    int i = blockIdx.x * blockDim.x + threadIdx.x;
    if (i >= B_ * 2 * H_) return;
    int b = i / (2 * H_), j = i % (2 * H_);
    float v = (j < H_) ? cf[b * H_ + j] : cb[b * H_ + j - H_];
    catc[i] = f32_to_bf16(v);
}

// dec init: c = pre, h = tanh(pre), ctxh = [bf16(0), bf16(h)]
__global__ void k_dec_init(const float* __restrict__ pre,
                           float* __restrict__ h, float* __restrict__ c,
                           unsigned short* __restrict__ ctxh)
{
    int idx = blockIdx.x * blockDim.x + threadIdx.x;
    if (idx >= B_ * H_) return;
    int b = idx >> 9, j = idx & (H_ - 1);
    float cv = pre[idx];
    float hv = tanhf(cv);
    c[idx] = cv; h[idx] = hv;
    ctxh[b * (2 * H_) + j] = 0;
    ctxh[b * (2 * H_) + H_ + j] = f32_to_bf16(hv);
}

// key_sb (S*B, H) -> key_bhs (B,H,S) ; val_sb -> val_bsh (B,S,H)
__global__ void k_kv_transpose(const float* __restrict__ key_sb,
                               const float* __restrict__ val_sb,
                               float* __restrict__ key_bhs,
                               float* __restrict__ val_bsh)
{
    int idx = blockIdx.x * blockDim.x + threadIdx.x;
    if (idx >= S_ * B_ * H_) return;
    int hh = idx & (H_ - 1);
    int r = idx >> 9;             // s*B + b
    int s = r >> 5, b = r & 31;
    key_bhs[((size_t)b * H_ + hh) * S_ + s] = key_sb[idx];
    val_bsh[((size_t)b * S_ + s) * H_ + hh] = val_sb[idx];
}

// attention for one decoder step; block per batch row, 512 threads
__global__ __launch_bounds__(512)
void k_attention(const float* __restrict__ h,
                 const float* __restrict__ key_bhs,
                 const float* __restrict__ val_bsh,
                 const int* __restrict__ src_lens,
                 unsigned short* __restrict__ ctxh,
                 unsigned short* __restrict__ concat_row)
{
    const int b = blockIdx.x;
    const int tid = threadIdx.x;
    __shared__ float sm[S_];

    if (tid < S_) {
        float acc = 0.0f;
        const float* kp = key_bhs + (size_t)b * H_ * S_ + tid;
        const float* hp = h + b * H_;
        for (int j = 0; j < H_; ++j) acc += hp[j] * kp[(size_t)j * S_];
        sm[tid] = acc;
    }
    __syncthreads();
    if (tid == 0) {
        int L = src_lens[b]; if (L > S_) L = S_; if (L < 1) L = 1;
        float mx = -1e30f;
        for (int s = 0; s < L; ++s) mx = fmaxf(mx, sm[s]);
        float sum = 0.0f;
        for (int s = 0; s < L; ++s) { float e = __expf(sm[s] - mx); sm[s] = e; sum += e; }
        float inv = 1.0f / fmaxf(sum, 1e-12f);
        for (int s = 0; s < L; ++s) sm[s] *= inv;
        for (int s = L; s < S_; ++s) sm[s] = 0.0f;
    }
    __syncthreads();
    if (tid < H_) {
        float acc = 0.0f;
        const float* vp = val_bsh + (size_t)b * S_ * H_ + tid;
        #pragma unroll 4
        for (int s = 0; s < S_; ++s) acc += sm[s] * vp[(size_t)s * H_];
        unsigned short cb16 = f32_to_bf16(acc);
        unsigned short hb16 = f32_to_bf16(h[b * H_ + tid]);
        ctxh[b * (2 * H_) + tid]        = cb16;
        ctxh[b * (2 * H_) + H_ + tid]   = hb16;
        concat_row[b * (2 * H_) + tid]      = cb16;
        concat_row[b * (2 * H_) + H_ + tid] = hb16;
    }
}

// log-softmax + NLL; one block per (t,b) row, masked-sum into out[0]
__global__ __launch_bounds__(256)
void k_nll(const float* __restrict__ logits, const int* __restrict__ tgt,
           float* __restrict__ out, int V)
{
    const int r = blockIdx.x;          // r = t*B + b
    const int t = r >> 5, b = r & 31;
    const float* row = logits + (size_t)r * V;
    __shared__ float red[256];

    float mx = -3.4e38f;
    for (int j = threadIdx.x; j < V; j += 256) mx = fmaxf(mx, row[j]);
    red[threadIdx.x] = mx; __syncthreads();
    for (int s = 128; s > 0; s >>= 1) {
        if (threadIdx.x < s) red[threadIdx.x] = fmaxf(red[threadIdx.x], red[threadIdx.x + s]);
        __syncthreads();
    }
    mx = red[0]; __syncthreads();

    float sum = 0.0f;
    for (int j = threadIdx.x; j < V; j += 256) sum += __expf(row[j] - mx);
    red[threadIdx.x] = sum; __syncthreads();
    for (int s = 128; s > 0; s >>= 1) {
        if (threadIdx.x < s) red[threadIdx.x] += red[threadIdx.x + s];
        __syncthreads();
    }
    if (threadIdx.x == 0) {
        int tok = tgt[b * T_ + t + 1];
        if (tok != 0) {
            float nll = mx + logf(red[0]) - row[tok];
            atomicAdd(out, nll);
        }
    }
}

// ---------------------------------------------------------------------------
// Host side
// ---------------------------------------------------------------------------
static inline dim3 g1(long long n, int t) { return dim3((unsigned)((n + t - 1) / t)); }

static void gemm(const unsigned short* A, int lda, const unsigned short* B, int ldb,
                 float* C, int ldc, const float* bias, int M, int N, int K,
                 hipStream_t s)
{
    dim3 grid(N / 256, M / 32);
    k_gemm_bf16<<<grid, 128, 0, s>>>(A, lda, B, ldb, C, ldc, bias, M, N, K);
}

extern "C" void kernel_launch(void* const* d_in, const int* in_sizes, int n_in,
                              void* d_out, int out_size, void* d_ws, size_t ws_size,
                              hipStream_t stream)
{
    (void)in_sizes; (void)n_in; (void)out_size; (void)ws_size;

    const int*   src_tokens = (const int*)  d_in[0];
    const int*   src_lens   = (const int*)  d_in[1];
    const int*   tgt_tokens = (const int*)  d_in[2];
    const float* src_emb    = (const float*)d_in[3];
    const float* tgt_emb    = (const float*)d_in[4];
    const float* enc_Wih_f  = (const float*)d_in[5];
    const float* enc_Whh_f  = (const float*)d_in[6];
    const float* enc_bih_f  = (const float*)d_in[7];
    const float* enc_bhh_f  = (const float*)d_in[8];
    const float* enc_Wih_b  = (const float*)d_in[9];
    const float* enc_Whh_b  = (const float*)d_in[10];
    const float* enc_bih_b  = (const float*)d_in[11];
    const float* enc_bhh_b  = (const float*)d_in[12];
    const float* dec_Wih    = (const float*)d_in[13];
    const float* dec_Whh    = (const float*)d_in[14];
    const float* dec_bih    = (const float*)d_in[15];
    const float* dec_bhh    = (const float*)d_in[16];
    const float* W_init     = (const float*)d_in[17];
    const float* b_init     = (const float*)d_in[18];
    const float* W_key      = (const float*)d_in[19];
    const float* b_key      = (const float*)d_in[20];
    const float* W_val      = (const float*)d_in[21];
    const float* b_val      = (const float*)d_in[22];
    const float* W_out      = (const float*)d_in[23];
    const float* b_out      = (const float*)d_in[24];
    float* out = (float*)d_out;

    // ---- workspace bump allocator -----------------------------------------
    char* wsp = (char*)d_ws;
    size_t off = 0;
    auto alloc = [&](size_t bytes) -> void* {
        void* p = wsp + off;
        off = (off + bytes + 255) & ~(size_t)255;
        return p;
    };
    auto aUS = [&](size_t n) { return (unsigned short*)alloc(n * 2); };
    auto aF  = [&](size_t n) { return (float*)alloc(n * 4); };

    unsigned short* Wihf_bf  = aUS((size_t)G_ * E_);
    unsigned short* Wihb_bf  = aUS((size_t)G_ * E_);
    unsigned short* Whhf_bf  = aUS((size_t)G_ * H_);
    unsigned short* Whhb_bf  = aUS((size_t)G_ * H_);
    unsigned short* Wx_bf    = aUS((size_t)G_ * E_);        // dec_Wih[:, 0:E]
    unsigned short* Wcomb_bf = aUS((size_t)G_ * 2 * H_);    // [dec_Wih[:,E:], dec_Whh]
    unsigned short* Winit_bf = aUS((size_t)H_ * 2 * H_);
    unsigned short* Wkey_bf  = aUS((size_t)H_ * 2 * H_);
    unsigned short* Wval_bf  = aUS((size_t)H_ * 2 * H_);
    unsigned short* Wout_bf  = aUS((size_t)V_ * 2 * H_);

    unsigned short* xsrc_bf  = aUS((size_t)S_ * B_ * E_);
    unsigned short* xtgt_bf  = aUS((size_t)TD_ * B_ * E_);
    float* XpreF  = aF((size_t)S_ * B_ * G_);
    float* XpreB  = aF((size_t)S_ * B_ * G_);
    float* XpreD  = aF((size_t)TD_ * B_ * G_);
    float* gates  = aF((size_t)B_ * G_);

    float* hF = aF(B_ * H_);  float* cF = aF(B_ * H_);
    float* hB = aF(B_ * H_);  float* cB = aF(B_ * H_);
    unsigned short* hF_bf = aUS(B_ * H_);
    unsigned short* hB_bf = aUS(B_ * H_);
    unsigned short* encout_bf = aUS((size_t)S_ * B_ * 2 * H_);
    unsigned short* catc_bf   = aUS((size_t)B_ * 2 * H_);
    float* decc_pre = aF(B_ * H_);
    float* hD = aF(B_ * H_);  float* cD = aF(B_ * H_);
    unsigned short* ctxh_bf   = aUS((size_t)B_ * 2 * H_);
    float* key_sb  = aF((size_t)S_ * B_ * H_);
    float* val_sb  = aF((size_t)S_ * B_ * H_);
    float* key_bhs = aF((size_t)B_ * H_ * S_);
    float* val_bsh = aF((size_t)B_ * S_ * H_);
    unsigned short* concat_bf = aUS((size_t)TD_ * B_ * 2 * H_);
    float* logits  = aF((size_t)TD_ * B_ * V_);

    // ---- init output accumulator ------------------------------------------
    k_zero_f32<<<1, 32, 0, stream>>>(out, 1);

    // ---- weight conversions to bf16 ---------------------------------------
    k_cvt_bf16<<<g1((size_t)G_*E_,256),256,0,stream>>>(enc_Wih_f, Wihf_bf, G_*E_);
    k_cvt_bf16<<<g1((size_t)G_*E_,256),256,0,stream>>>(enc_Wih_b, Wihb_bf, G_*E_);
    k_cvt_bf16<<<g1((size_t)G_*H_,256),256,0,stream>>>(enc_Whh_f, Whhf_bf, G_*H_);
    k_cvt_bf16<<<g1((size_t)G_*H_,256),256,0,stream>>>(enc_Whh_b, Whhb_bf, G_*H_);
    k_cvt_bf16<<<g1((size_t)H_*2*H_,256),256,0,stream>>>(W_init, Winit_bf, H_*2*H_);
    k_cvt_bf16<<<g1((size_t)H_*2*H_,256),256,0,stream>>>(W_key,  Wkey_bf,  H_*2*H_);
    k_cvt_bf16<<<g1((size_t)H_*2*H_,256),256,0,stream>>>(W_val,  Wval_bf,  H_*2*H_);
    k_cvt_bf16<<<g1((size_t)V_*2*H_,256),256,0,stream>>>(W_out,  Wout_bf,  V_*2*H_);
    // dec_Wih (G, E+H) split: x-slice and ctx-slice; Wcomb = [Wih_ctx | Whh]
    k_slice_bf16<<<g1((size_t)G_*E_,256),256,0,stream>>>(dec_Wih, E_+H_, 0,  Wx_bf,   E_,   0,  G_, E_);
    k_slice_bf16<<<g1((size_t)G_*H_,256),256,0,stream>>>(dec_Wih, E_+H_, E_, Wcomb_bf,2*H_, 0,  G_, H_);
    k_slice_bf16<<<g1((size_t)G_*H_,256),256,0,stream>>>(dec_Whh, H_,    0,  Wcomb_bf,2*H_, H_, G_, H_);

    // ---- embeddings -------------------------------------------------------
    // src: row (s,b) token = src_tokens[s*B + b]
    k_embed<<<g1((size_t)S_*B_*E_,256),256,0,stream>>>(src_tokens, B_, 1, src_emb, xsrc_bf, S_, B_, E_);
    // tgt: row (t,b) token = tgt_tokens[b*T + t], t < T-1
    k_embed<<<g1((size_t)TD_*B_*E_,256),256,0,stream>>>(tgt_tokens, 1, T_, tgt_emb, xtgt_bf, TD_, B_, E_);

    // ---- hoisted input GEMMs (all timesteps at once) ----------------------
    gemm(xsrc_bf, E_, Wihf_bf, E_, XpreF, G_, nullptr, S_*B_, G_, E_, stream);
    gemm(xsrc_bf, E_, Wihb_bf, E_, XpreB, G_, nullptr, S_*B_, G_, E_, stream);
    gemm(xtgt_bf, E_, Wx_bf,   E_, XpreD, G_, nullptr, TD_*B_, G_, E_, stream);

    // ---- encoder forward scan ---------------------------------------------
    k_zero_f32<<<g1(B_*H_,256),256,0,stream>>>(hF, B_*H_);
    k_zero_f32<<<g1(B_*H_,256),256,0,stream>>>(cF, B_*H_);
    k_zero_us <<<g1(B_*H_,256),256,0,stream>>>(hF_bf, B_*H_);
    for (int t = 0; t < S_; ++t) {
        gemm(hF_bf, H_, Whhf_bf, H_, gates, G_, nullptr, B_, G_, H_, stream);
        k_lstm_enc_step<<<g1(B_*H_,256),256,0,stream>>>(
            gates, XpreF + (size_t)t*B_*G_, enc_bih_f, enc_bhh_f,
            hF, cF, hF_bf, encout_bf, src_lens, t, 0);
    }
    // ---- encoder backward scan (chronologically reversed) ------------------
    k_zero_f32<<<g1(B_*H_,256),256,0,stream>>>(hB, B_*H_);
    k_zero_f32<<<g1(B_*H_,256),256,0,stream>>>(cB, B_*H_);
    k_zero_us <<<g1(B_*H_,256),256,0,stream>>>(hB_bf, B_*H_);
    for (int t = S_ - 1; t >= 0; --t) {
        gemm(hB_bf, H_, Whhb_bf, H_, gates, G_, nullptr, B_, G_, H_, stream);
        k_lstm_enc_step<<<g1(B_*H_,256),256,0,stream>>>(
            gates, XpreB + (size_t)t*B_*G_, enc_bih_b, enc_bhh_b,
            hB, cB, hB_bf, encout_bf, src_lens, t, H_);
    }

    // ---- attention keys/values + decoder init -----------------------------
    gemm(encout_bf, 2*H_, Wkey_bf, 2*H_, key_sb, H_, b_key, S_*B_, H_, 2*H_, stream);
    gemm(encout_bf, 2*H_, Wval_bf, 2*H_, val_sb, H_, b_val, S_*B_, H_, 2*H_, stream);
    k_kv_transpose<<<g1((size_t)S_*B_*H_,256),256,0,stream>>>(key_sb, val_sb, key_bhs, val_bsh);

    k_concat_c<<<g1(B_*2*H_,256),256,0,stream>>>(cF, cB, catc_bf);
    gemm(catc_bf, 2*H_, Winit_bf, 2*H_, decc_pre, H_, b_init, B_, H_, 2*H_, stream);
    k_dec_init<<<g1(B_*H_,256),256,0,stream>>>(decc_pre, hD, cD, ctxh_bf);

    // ---- decoder scan ------------------------------------------------------
    for (int t = 0; t < TD_; ++t) {
        gemm(ctxh_bf, 2*H_, Wcomb_bf, 2*H_, gates, G_, nullptr, B_, G_, 2*H_, stream);
        k_lstm_dec_step<<<g1(B_*H_,256),256,0,stream>>>(
            gates, XpreD + (size_t)t*B_*G_, dec_bih, dec_bhh, hD, cD);
        k_attention<<<B_, 512, 0, stream>>>(
            hD, key_bhs, val_bsh, src_lens, ctxh_bf,
            concat_bf + (size_t)t*B_*2*H_);
    }

    // ---- batched output projection (dominant GEMM) + NLL -------------------
    gemm(concat_bf, 2*H_, Wout_bf, 2*H_, logits, V_, b_out, TD_*B_, V_, 2*H_, stream);
    k_nll<<<TD_*B_, 256, 0, stream>>>(logits, tgt_tokens, out, V_);
}
